// PointConvAttn_29798483100372
// MI455X (gfx1250) — compile-verified, matrix operand order checked by reading, not run on hardware
//
#include <hip/hip_runtime.h>
#include <cstddef>
#include <cstdint>

// ---------------------------------------------------------------------------
// Problem constants (from reference): B=4, N=2048, D=3, CIN=64, CMID=16, H=4,
// WH=32, FH=512, COUT=64, FIN=4096, NB=16.  Q = B*N = 8192 query points.
// ---------------------------------------------------------------------------
#define QTOT   8192
#define NPTS   2048
#define NB     16
#define CIN    64
#define FIN    4096
#define FH     512
#define COUT   64

typedef __attribute__((ext_vector_type(16))) __bf16        v16bf;
typedef __attribute__((ext_vector_type(8)))  float         v8f;
typedef __attribute__((ext_vector_type(4)))  unsigned int  v4u;
typedef __attribute__((ext_vector_type(8)))  unsigned int  v8u;

// ---------------------------------------------------------------------------
// Workspace layout (bytes)
// ---------------------------------------------------------------------------
#define OFS_IDX  ((size_t)0)                      // int   [8192*16]        512 KB
#define OFS_MT   ((size_t)524288)                 // bf16  [8192*64*16]      16 MB  (m^T per point)
#define OFS_NFT  ((size_t)17301504)               // bf16  [8192*64*16]      16 MB  (nf^T per point)
#define OFS_E    ((size_t)34078720)               // bf16  [8192*4096]       64 MB
#define OFS_H    ((size_t)101187584)              // bf16  [8192*512]         8 MB
#define OFS_W1T  ((size_t)109576192)              // bf16  [512*4096]         4 MB
#define OFS_W2T  ((size_t)113770496)              // bf16  [64*512]          64 KB

// ---------------------------------------------------------------------------
// Kernel 0a/0b: transpose fc weights into bf16 [N][K] so a WMMA B-operand
// lane reads 16 contiguous bf16 (32B) for its column.
// ---------------------------------------------------------------------------
__global__ __launch_bounds__(256) void prep_w1t_kernel(const float* __restrict__ w1,
                                                       __bf16* __restrict__ w1t) {
    int i = blockIdx.x * 256 + threadIdx.x;          // over 512*4096
    if (i < FH * FIN) {
        int j = i >> 12;                             // output col (0..511)
        int k = i & (FIN - 1);                       // K (0..4095)
        w1t[i] = (__bf16)w1[k * FH + j];
    }
}

__global__ __launch_bounds__(256) void prep_w2t_kernel(const float* __restrict__ w2,
                                                       __bf16* __restrict__ w2t) {
    int i = blockIdx.x * 256 + threadIdx.x;          // over 64*512
    if (i < COUT * FH) {
        int j = i >> 9;                              // output col (0..63)
        int k = i & (FH - 1);                        // K (0..511)
        w2t[i] = (__bf16)w2[k * COUT + j];
    }
}

// ---------------------------------------------------------------------------
// Kernel 1: kNN.  One wave (32 lanes) per query point.  Distances staged in
// LDS (2048 f32 per wave), then 16 rounds of argmin with wave32 butterfly
// reductions.  Invalid candidates get +1e30.
// ---------------------------------------------------------------------------
__global__ __launch_bounds__(128) void knn_kernel(const float* __restrict__ keys,
                                                  const float* __restrict__ points,
                                                  const unsigned char* __restrict__ valid,
                                                  int* __restrict__ nidx) {
    __shared__ float sd[4][NPTS];
    const int wid = threadIdx.x >> 5;
    const int l   = threadIdx.x & 31;
    const int q   = blockIdx.x * 4 + wid;
    const int b   = q >> 11;
    const int k   = q & (NPTS - 1);

    const float kx = keys[(b * NPTS + k) * 3 + 0];
    const float ky = keys[(b * NPTS + k) * 3 + 1];
    const float kz = keys[(b * NPTS + k) * 3 + 2];
    const float* pb = points + (size_t)b * NPTS * 3;
    const unsigned char* vrow = valid + ((size_t)(b * NPTS + k)) * NPTS;

    #pragma unroll 4
    for (int t = 0; t < NPTS / 32; ++t) {
        int j = t * 32 + l;
        float dx = pb[j * 3 + 0] - kx;
        float dy = pb[j * 3 + 1] - ky;
        float dz = pb[j * 3 + 2] - kz;
        float d = dx * dx + dy * dy + dz * dz;
        if (!vrow[j]) d += 1.0e30f;
        sd[wid][j] = d;
    }

    for (int r = 0; r < NB; ++r) {
        float mv = 3.0e38f;
        int   mj = -1;
        for (int t = 0; t < NPTS / 32; ++t) {
            int j = t * 32 + l;
            float d = sd[wid][j];
            if (d < mv) { mv = d; mj = j; }
        }
        #pragma unroll
        for (int off = 16; off >= 1; off >>= 1) {
            float ov = __shfl_xor(mv, off, 32);
            int   oj = __shfl_xor(mj, off, 32);
            if (ov < mv || (ov == mv && (unsigned)oj < (unsigned)mj)) { mv = ov; mj = oj; }
        }
        if (l == 0) {
            nidx[q * NB + r] = mj;
            sd[wid][mj] = 3.4e38f;
        }
    }
}

// ---------------------------------------------------------------------------
// Kernel 2: per-point edge computation.
// ---------------------------------------------------------------------------
__global__ __launch_bounds__(128) void edge_kernel(const float* __restrict__ keys,
                                                   const float* __restrict__ points,
                                                   const float* __restrict__ feats,
                                                   const unsigned char* __restrict__ valid,
                                                   const int* __restrict__ nidx,
                                                   const float* __restrict__ wc_w1,
                                                   const float* __restrict__ wc_b1,
                                                   const float* __restrict__ wc_w2,
                                                   const float* __restrict__ wc_b2,
                                                   const float* __restrict__ at_w1,
                                                   const float* __restrict__ at_b1,
                                                   const float* __restrict__ at_w2,
                                                   const float* __restrict__ at_b2,
                                                   __bf16* __restrict__ mT,
                                                   __bf16* __restrict__ nfT) {
    __shared__ float s_nf[4][NB][CIN];
    __shared__ float s_hid[4][NB][64];
    __shared__ float s_attn[4][NB][4];
    __shared__ float s_hw[4][NB][32];
    __shared__ float s_m[4][NB][16];
    __shared__ int   s_idx[4][NB];
    __shared__ float s_vf[4][NB];

    const int wid = threadIdx.x >> 5;
    const int l   = threadIdx.x & 31;
    const int q   = blockIdx.x * 4 + wid;
    const int b   = q >> 11;
    const int k   = q & (NPTS - 1);

    if (l < NB) {
        int j = nidx[q * NB + l];
        s_idx[wid][l] = j;
        s_vf[wid][l] = valid[((size_t)(b * NPTS + k)) * NPTS + j] ? 1.0f : 0.0f;
    }
    __syncthreads();

    for (int n = 0; n < NB; ++n) {
        const float* f = feats + ((size_t)(b * NPTS + s_idx[wid][n])) * CIN;
        s_nf[wid][n][l]      = f[l];
        s_nf[wid][n][l + 32] = f[l + 32];
    }

    const int hd0 = l, hd1 = l + 32;
    float sc0 = at_b1[hd0], sc1 = at_b1[hd1];
    const float* cf = feats + ((size_t)(b * NPTS + k)) * CIN;
    for (int c = 0; c < CIN; ++c) {
        float fv = cf[c];
        sc0 += fv * at_w1[c * 64 + hd0];
        sc1 += fv * at_w1[c * 64 + hd1];
    }
    __syncthreads();

    for (int n = 0; n < NB; ++n) {
        float h0 = sc0, h1 = sc1;
        for (int c = 0; c < CIN; ++c) {
            float nf = s_nf[wid][n][c];
            h0 += nf * at_w1[(64 + c) * 64 + hd0];
            h1 += nf * at_w1[(64 + c) * 64 + hd1];
        }
        s_hid[wid][n][hd0] = fmaxf(h0, 0.0f);
        s_hid[wid][n][hd1] = fmaxf(h1, 0.0f);
    }
    __syncthreads();

    {
        const int nn = l & 15;
        const int hh = l >> 4;
        float r0 = at_b2[hh], r1 = at_b2[hh + 2];
        for (int hd = 0; hd < 64; ++hd) {
            float hv = s_hid[wid][nn][hd];
            r0 += hv * at_w2[hd * 4 + hh];
            r1 += hv * at_w2[hd * 4 + hh + 2];
        }
        float m0 = r0, m1 = r1;
        #pragma unroll
        for (int off = 8; off >= 1; off >>= 1) {
            m0 = fmaxf(m0, __shfl_xor(m0, off, 16));
            m1 = fmaxf(m1, __shfl_xor(m1, off, 16));
        }
        float e0 = __expf(r0 - m0), e1 = __expf(r1 - m1);
        float su0 = e0, su1 = e1;
        #pragma unroll
        for (int off = 8; off >= 1; off >>= 1) {
            su0 += __shfl_xor(su0, off, 16);
            su1 += __shfl_xor(su1, off, 16);
        }
        s_attn[wid][nn][hh]     = e0 / su0;
        s_attn[wid][nn][hh + 2] = e1 / su1;
    }
    __syncthreads();

    {
        const float kx = keys[(b * NPTS + k) * 3 + 0];
        const float ky = keys[(b * NPTS + k) * 3 + 1];
        const float kz = keys[(b * NPTS + k) * 3 + 2];
        for (int n = 0; n < NB; ++n) {
            int j = s_idx[wid][n];
            float dx = points[(b * NPTS + j) * 3 + 0] - kx;
            float dy = points[(b * NPTS + j) * 3 + 1] - ky;
            float dz = points[(b * NPTS + j) * 3 + 2] - kz;
            float hw = wc_b1[l] + dx * wc_w1[l] + dy * wc_w1[32 + l] + dz * wc_w1[64 + l];
            s_hw[wid][n][l] = fmaxf(hw, 0.0f);
        }
    }
    __syncthreads();

    {
        const int cm = l & 15;
        for (int it = 0; it < 8; ++it) {
            int n2 = (l >> 4) + it * 2;
            float mv = wc_b2[cm];
            for (int wh = 0; wh < 32; ++wh) mv += s_hw[wid][n2][wh] * wc_w2[wh * 16 + cm];
            s_m[wid][n2][cm] = mv;
        }
    }
    __syncthreads();

    __bf16* mTo  = mT  + (size_t)q * (64 * NB);
    __bf16* nfTo = nfT + (size_t)q * (CIN * NB);
    for (int n = 0; n < NB; ++n) {
        float vf = s_vf[wid][n];
        int mr0 = l, mr1 = l + 32;
        mTo[mr0 * NB + n] = (__bf16)(s_attn[wid][n][mr0 >> 4] * s_m[wid][n][mr0 & 15] * vf);
        mTo[mr1 * NB + n] = (__bf16)(s_attn[wid][n][mr1 >> 4] * s_m[wid][n][mr1 & 15] * vf);
        nfTo[l        * NB + n] = (__bf16)(s_nf[wid][n][l]      * vf);
        nfTo[(l + 32) * NB + n] = (__bf16)(s_nf[wid][n][l + 32] * vf);
    }
}

// ---------------------------------------------------------------------------
// Kernel 3: per-point e = m^T (64x16) @ nf (16x64) via WMMA bf16, K padded
// to 32 (lanes 16-31 carry the zero K-half).  16 v_wmma per point.
// ---------------------------------------------------------------------------
__global__ __launch_bounds__(128) void e_kernel(const __bf16* __restrict__ mT,
                                                const __bf16* __restrict__ nfT,
                                                __bf16* __restrict__ eb) {
    const int wid = threadIdx.x >> 5;
    const int l   = threadIdx.x & 31;
    const int q   = blockIdx.x * 4 + wid;
    const int half = l >> 4;
    const int lr   = l & 15;

    const __bf16* a0 = mT  + (size_t)q * (64 * NB);
    const __bf16* b0 = nfT + (size_t)q * (CIN * NB);
    __bf16*       eo = eb  + (size_t)q * FIN;

    v16bf z = {};
    v16bf A[4], Bt[4];
    #pragma unroll
    for (int t = 0; t < 4; ++t) {
        if (half == 0) {
            A[t]  = *(const v16bf*)(a0 + (t * 16 + lr) * NB);
            Bt[t] = *(const v16bf*)(b0 + (t * 16 + lr) * NB);
        } else {
            A[t]  = z;
            Bt[t] = z;
        }
    }

    #pragma unroll
    for (int mt = 0; mt < 4; ++mt) {
        #pragma unroll
        for (int nt = 0; nt < 4; ++nt) {
            v8f acc = {};
            acc = __builtin_amdgcn_wmma_f32_16x16x32_bf16(false, A[mt], false, Bt[nt],
                                                          (short)0, acc, false, false);
            #pragma unroll
            for (int r = 0; r < 8; ++r) {
                int M = r + 8 * half;
                eo[(mt * 16 + M) * CIN + nt * 16 + lr] = (__bf16)acc[r];
            }
        }
    }
}

// ---------------------------------------------------------------------------
// Kernel 4: GEMM1  h = relu(e @ fc_w1 + b1)   M=8192 N=512 K=4096.
// Block = 8 waves = 128(M) x 64(N) tile.  The 64-col x 128-K B-slab (16 KB)
// is DMA'd into LDS by the Tensor Data Mover, DOUBLE-BUFFERED: wave 0 issues
// the DMA for chunk i+1 into the other slab, then s_wait_tensorcnt(1) — TDM
// ops from one wave complete in order, so TENSORcnt<=1 means chunk i landed
// while chunk i+1 streams in the background.
//
// TDM D# (CDNA5 ISA):
//   group0: count=1 | lds_addr | 57b global_addr | type=2
//   group1: data_size=2B, tensor_dim0=tile_dim0=128, tensor_dim1=tile_dim1=64,
//           tensor_dim0_stride=4096, pad_enable, pad_interval=64dw (one 256B
//           row), pad_amount=8dw (32B) -> LDS row stride 288B (32B-aligned
//           v16bf reads, rows 8 banks apart)
//   group2/3: benign 2D fill.
// ---------------------------------------------------------------------------
#define G1_KCHUNK 128
#define G1_ROWB   288          // 256B row + 32B pad
#define G1_SLAB   (64 * G1_ROWB)

__global__ __launch_bounds__(256) void gemm1_kernel(const __bf16* __restrict__ eb,
                                                    const __bf16* __restrict__ w1t,
                                                    const float* __restrict__ b1,
                                                    __bf16* __restrict__ hb) {
    __shared__ __attribute__((aligned(128))) unsigned char sB[2][G1_SLAB];

    const int l    = threadIdx.x & 31;
    const int wid  = threadIdx.x >> 5;
    const int mblk = blockIdx.x >> 3;            // 64 M-blocks of 128 rows
    const int ng   = blockIdx.x & 7;             // 8 N-groups of 64 cols
    const int half = l >> 4;
    const int lr   = l & 15;
    const int base_col = ng * 64;
    const int row0 = mblk * 128 + wid * 16;

    const __bf16* Ap = eb + ((size_t)(row0 + lr)) * FIN + half * 16;
    const __bf16* Bg = w1t + (size_t)base_col * FIN;

    // chunk-invariant descriptor words
    const unsigned int g1w0 = (1u << 16)          // data_size = 2 bytes
                            | (1u << 20)          // pad_enable
                            | (5u << 22)          // pad_interval: 64 dwords (one row)
                            | (7u << 25);         // pad_amount: 8 dwords (32B)
    const v8u g1 = { g1w0,
                     (unsigned)G1_KCHUNK << 16,   // tensor_dim0[15:0] @ bits 63:48
                     64u << 16,                   // dim0[31:16]=0 | tensor_dim1[15:0]
                     (unsigned)G1_KCHUNK << 16,   // dim1[31:16]=0 | tile_dim0
                     64u,                         // tile_dim1 | tile_dim2=0
                     (unsigned)FIN,               // tensor_dim0_stride[31:0]
                     0u, 0u };
    const v4u g2 = { 1u, 1u, 0u, 0u };
    const v4u g3 = { 0u, 0u, 0u, 0u };
    const unsigned int lds_base = (unsigned int)(uintptr_t)&sB[0][0];

    v8f zf = {};
    v8f acc[4] = {zf, zf, zf, zf};

    const int NCHUNK = FIN / G1_KCHUNK;          // 32

    // prologue: wave 0 kicks off chunk 0 into slab 0
    if (wid == 0) {
        unsigned long long ga = (unsigned long long)(uintptr_t)Bg;
        v4u g0 = { 1u, lds_base, (unsigned int)ga,
                   (unsigned int)((ga >> 32) & 0x01FFFFFFu) | 0x80000000u };
        asm volatile("tensor_load_to_lds %0, %1, %2, %3"
                     :: "s"(g0), "s"(g1), "s"(g2), "s"(g3) : "memory");
    }

    for (int c = 0; c < NCHUNK; ++c) {
        const int k0  = c * G1_KCHUNK;
        const int buf = c & 1;
        if (wid == 0) {
            if (c + 1 < NCHUNK) {                // issue next chunk into other slab
                unsigned long long ga =
                    (unsigned long long)(uintptr_t)(Bg + k0 + G1_KCHUNK);
                v4u g0 = { 1u, lds_base + (unsigned)((buf ^ 1) * G1_SLAB),
                           (unsigned int)ga,
                           (unsigned int)((ga >> 32) & 0x01FFFFFFu) | 0x80000000u };
                asm volatile("tensor_load_to_lds %0, %1, %2, %3"
                             :: "s"(g0), "s"(g1), "s"(g2), "s"(g3) : "memory");
                __builtin_amdgcn_s_wait_tensorcnt(1);   // chunk c landed (in-order)
            } else {
                __builtin_amdgcn_s_wait_tensorcnt(0);   // last chunk landed
            }
        }
        __syncthreads();                         // publish slab buf

        #pragma unroll
        for (int kk = 0; kk < G1_KCHUNK; kk += 32) {
            __builtin_prefetch(Ap + k0 + kk + 256, 0, 1);
            v16bf a = *(const v16bf*)(Ap + k0 + kk);
            #pragma unroll
            for (int s = 0; s < 4; ++s) {
                const unsigned char* bp =
                    sB[buf] + (s * 16 + lr) * G1_ROWB + (kk + half * 16) * 2;
                v16bf bm = *(const v16bf*)bp;
                acc[s] = __builtin_amdgcn_wmma_f32_16x16x32_bf16(false, a, false, bm,
                                                                 (short)0, acc[s],
                                                                 false, false);
            }
        }
        __syncthreads();                         // slab buf consumed; safe to
                                                 // overwrite at issue of c+2
    }

    #pragma unroll
    for (int s = 0; s < 4; ++s) {
        #pragma unroll
        for (int r = 0; r < 8; ++r) {
            int row = row0 + r + 8 * half;
            int col = base_col + s * 16 + lr;
            float v = acc[s][r] + b1[col];
            hb[(size_t)row * FH + col] = (__bf16)fmaxf(v, 0.0f);
        }
    }
}

// ---------------------------------------------------------------------------
// Kernel 5: GEMM2  out = h @ fc_w2 + b2   M=8192 N=64 K=512 (bf16 WMMA, f32 out)
// ---------------------------------------------------------------------------
__global__ __launch_bounds__(256) void gemm2_kernel(const __bf16* __restrict__ hb,
                                                    const __bf16* __restrict__ w2t,
                                                    const float* __restrict__ b2,
                                                    float* __restrict__ out) {
    const int l    = threadIdx.x & 31;
    const int mtil = blockIdx.x * 8 + (threadIdx.x >> 5);
    const int half = l >> 4;
    const int lr   = l & 15;

    const __bf16* Ap = hb + ((size_t)(mtil * 16 + lr)) * FH + half * 16;
    const __bf16* Bp[4];
    #pragma unroll
    for (int s = 0; s < 4; ++s)
        Bp[s] = w2t + ((size_t)(s * 16 + lr)) * FH + half * 16;

    v8f zf = {};
    v8f acc[4] = {zf, zf, zf, zf};

    #pragma unroll 4
    for (int k = 0; k < FH; k += 32) {
        v16bf a = *(const v16bf*)(Ap + k);
        #pragma unroll
        for (int s = 0; s < 4; ++s) {
            v16bf bm = *(const v16bf*)(Bp[s] + k);
            acc[s] = __builtin_amdgcn_wmma_f32_16x16x32_bf16(false, a, false, bm,
                                                             (short)0, acc[s], false, false);
        }
    }

    #pragma unroll
    for (int s = 0; s < 4; ++s) {
        #pragma unroll
        for (int r = 0; r < 8; ++r) {
            int row = mtil * 16 + r + 8 * half;
            int col = s * 16 + lr;
            out[(size_t)row * COUT + col] = acc[s][r] + b2[col];
        }
    }
}

// ---------------------------------------------------------------------------
// Host launcher
// ---------------------------------------------------------------------------
extern "C" void kernel_launch(void* const* d_in, const int* in_sizes, int n_in,
                              void* d_out, int out_size, void* d_ws, size_t ws_size,
                              hipStream_t stream) {
    (void)in_sizes; (void)n_in; (void)out_size; (void)ws_size;

    const float* keys   = (const float*)d_in[0];
    const float* points = (const float*)d_in[1];
    const float* feats  = (const float*)d_in[2];
    const unsigned char* valid = (const unsigned char*)d_in[3];
    const float* wc_w1 = (const float*)d_in[4];
    const float* wc_b1 = (const float*)d_in[5];
    const float* wc_w2 = (const float*)d_in[6];
    const float* wc_b2 = (const float*)d_in[7];
    const float* at_w1 = (const float*)d_in[8];
    const float* at_b1 = (const float*)d_in[9];
    const float* at_w2 = (const float*)d_in[10];
    const float* at_b2 = (const float*)d_in[11];
    const float* fc_w1 = (const float*)d_in[12];
    const float* fc_b1 = (const float*)d_in[13];
    const float* fc_w2 = (const float*)d_in[14];
    const float* fc_b2 = (const float*)d_in[15];
    float* out = (float*)d_out;

    char* ws = (char*)d_ws;
    int*    nidx = (int*)   (ws + OFS_IDX);
    __bf16* mT   = (__bf16*)(ws + OFS_MT);
    __bf16* nfT  = (__bf16*)(ws + OFS_NFT);
    __bf16* eb   = (__bf16*)(ws + OFS_E);
    __bf16* hb   = (__bf16*)(ws + OFS_H);
    __bf16* w1t  = (__bf16*)(ws + OFS_W1T);
    __bf16* w2t  = (__bf16*)(ws + OFS_W2T);

    prep_w1t_kernel<<<(FH * FIN + 255) / 256, 256, 0, stream>>>(fc_w1, w1t);
    prep_w2t_kernel<<<(COUT * FH + 255) / 256, 256, 0, stream>>>(fc_w2, w2t);

    knn_kernel<<<QTOT / 4, 128, 0, stream>>>(keys, points, valid, nidx);

    edge_kernel<<<QTOT / 4, 128, 0, stream>>>(keys, points, feats, valid, nidx,
                                              wc_w1, wc_b1, wc_w2, wc_b2,
                                              at_w1, at_b1, at_w2, at_b2,
                                              mT, nfT);

    e_kernel<<<QTOT / 4, 128, 0, stream>>>(mT, nfT, eb);

    // gemm1: 64 M-blocks (128 rows) x 8 N-groups = 512 blocks of 8 waves
    gemm1_kernel<<<512, 256, 0, stream>>>(eb, w1t, fc_b1, hb);
    // gemm2: 512 M-tiles, 1 wave each = 64 blocks of 8 waves
    gemm2_kernel<<<64, 256, 0, stream>>>(hb, w2t, fc_b2, out);
}
